// RN_30116310680400
// MI455X (gfx1250) — matrix operand order repaired and checked by reading, not compiled, not run on hardware
//
#include <hip/hip_runtime.h>
#include <hip/hip_bf16.h>

// Sizes fixed by the reference
#define Bn   2
#define Dd   64
#define Nn   512
#define Hh   256        // hidden = 4*D
#define OUTD 64
#define LDS_PITCH 264   // 256 f16 + 8 f16 pad to stagger LDS banks
#define CPB  4          // context rows (c) processed per block
#define ITILES 8        // i-range per block = 64 rows (4 strips x 16)

typedef _Float16 v8h  __attribute__((ext_vector_type(8)));
typedef _Float16 v16h __attribute__((ext_vector_type(16)));
typedef float    v8f  __attribute__((ext_vector_type(8)));

union V16 { v16h v; v8h h[2]; };

// ---------------------------------------------------------------------------
// Pre-pass 1: hi = x@w1[:D] + b1, hc = x@w1[D:]   (stored f16, [b*N+n][H])
// input is [B][D][N] channels-first; x[b,n,d] = input[b][d][n].
// ---------------------------------------------------------------------------
__global__ __launch_bounds__(Hh)
void prep_hihc(const float* __restrict__ input,
               const float* __restrict__ w1,
               const float* __restrict__ b1,
               _Float16* __restrict__ hi,
               _Float16* __restrict__ hc) {
    const int bn = blockIdx.x;          // 0..B*N-1
    const int b  = bn >> 9;             // /512
    const int n  = bn & (Nn - 1);
    const int h  = threadIdx.x;         // 0..255
    const float* xp   = input + (size_t)b * Dd * Nn + n;   // stride N per d
    const float* w1lo = w1 + h;                            // w1[d][h], stride H
    const float* w1hi = w1 + (size_t)Dd * Hh + h;
    float a1 = 0.f, a2 = 0.f;
#pragma unroll 4
    for (int d = 0; d < Dd; ++d) {
        const float xv = xp[(size_t)d * Nn];
        a1 = fmaf(xv, w1lo[(size_t)d * Hh], a1);
        a2 = fmaf(xv, w1hi[(size_t)d * Hh], a2);
    }
    const size_t o = (size_t)bn * Hh + h;
    hi[o] = (_Float16)(a1 + b1[h]);     // bias folded into the "i" half
    hc[o] = (_Float16)a2;
}

// ---------------------------------------------------------------------------
// Pre-pass 2: w2t[n][k] = (f16) w2[k][n]   (unpadded, [256][256])
// ---------------------------------------------------------------------------
__global__ __launch_bounds__(Hh)
void prep_w2t(const float* __restrict__ w2,
              _Float16* __restrict__ w2t) {
    const int n = blockIdx.x;
    const int k = threadIdx.x;
    w2t[(size_t)n * Hh + k] = (_Float16)w2[(size_t)k * Hh + n];
}

// ---------------------------------------------------------------------------
// Main pair-GEMM: partial[block][col] = Sigma_rows relu( relu(hi[i]+hc[c]) @ w2 + b2 )
// grid = (8 i-tiles, 128 c-blocks, 2 b), 256 threads = 8 waves.
// Wave layout: strip = wave>>1 (4 row strips of 16 i-rows), ch = wave&1
// (column half, 128 cols = 8 WMMA tiles). Each block loops over CPB=4 c's.
// Per wave: CPB * 8 ksteps * 8 tiles = 256 x V_WMMA_F32_16X16X32_F16.
// ---------------------------------------------------------------------------
__global__ __launch_bounds__(256)
void pair_gemm(const _Float16* __restrict__ hi,
               const _Float16* __restrict__ hc,
               const _Float16* __restrict__ w2t,
               const float* __restrict__ b2,
               float* __restrict__ partials) {
    extern __shared__ char smem[];
    _Float16* w2s  = (_Float16*)smem;                                   // [256][264]
    float*    wsum = (float*)(smem + (size_t)Hh * LDS_PITCH * 2);       // [8][128]

    const int tid   = threadIdx.x;
    const int lane  = tid & 31;
    const int wave  = tid >> 5;
    const int strip = wave >> 1;     // 0..3 row strip
    const int ch    = wave & 1;      // 0..1 column half
    const int row   = lane & 15;     // M index within tile / N index for B
    const int hs    = lane >> 4;     // lane-half select per ISA layouts
    const int b     = blockIdx.z;
    const int cb    = blockIdx.y;    // c block: covers c = cb*CPB .. +CPB-1
    const int it    = blockIdx.x;

    // Stage w2^T into LDS with row padding (coalesced global reads).
    for (int base = tid * 8; base < Hh * Hh; base += 256 * 8) {
        const int n = base >> 8;
        const int k = base & (Hh - 1);
        *(v8h*)&w2s[(size_t)n * LDS_PITCH + k] = *(const v8h*)&w2t[base];
    }
    __syncthreads();

    const int i0 = it * 64 + strip * 16;
    const _Float16* hiRow = hi + ((size_t)(b * Nn + i0 + row)) * Hh;

    const v8h zero8 = {};
    float colAcc[8];                 // running per-column sums across c-loop
#pragma unroll
    for (int t = 0; t < 8; ++t) colAcc[t] = 0.f;

    for (int cc = 0; cc < CPB; ++cc) {
        const int c = cb * CPB + cc;
        const _Float16* hcRow = hc + ((size_t)(b * Nn + c)) * Hh;

        v8f acc[8];
#pragma unroll
        for (int t = 0; t < 8; ++t)
            acc[t] = v8f{0.f,0.f,0.f,0.f,0.f,0.f,0.f,0.f};

        for (int s = 0; s < 8; ++s) {
            const int kb = s * 32;
            // --- A fragment (16x32 f16): VGPR0-3 <- K = kb+hs*8+{0..7},
            //                             VGPR4-7 <- K = kb+16+hs*8+{0..7}
            v8h x0 = *(const v8h*)(hiRow + kb + hs * 8);
            v8h x1 = *(const v8h*)(hiRow + kb + 16 + hs * 8);
            v8h y0 = *(const v8h*)(hcRow + kb + hs * 8);
            v8h y1 = *(const v8h*)(hcRow + kb + 16 + hs * 8);
            V16 a;
            a.h[0] = __builtin_elementwise_max(x0 + y0, zero8); // v_pk_max_num_f16
            a.h[1] = __builtin_elementwise_max(x1 + y1, zero8);

#pragma unroll
            for (int t = 0; t < 8; ++t) {
                // --- B fragment (32x16 f16): elems 0-7 hold K=kb+hs*8+{0..7} of
                // column n, elems 8-15 hold K=kb+16+hs*8+{0..7} (same as A).
                const int n = ch * 128 + t * 16 + row;
                const _Float16* bp =
                    &w2s[(size_t)n * LDS_PITCH + kb + hs * 8];
                V16 bm;
                bm.h[0] = *(const v8h*)bp;         // ds_load_b128
                bm.h[1] = *(const v8h*)(bp + 16);  // ds_load_b128
                acc[t] = __builtin_amdgcn_wmma_f32_16x16x32_f16(
                    false, a.v, false, bm.v, (short)0, acc[t], false, false);
            }
        }

        // Fold this c's 16 rows into the running column sums: +b2, relu, sum.
#pragma unroll
        for (int t = 0; t < 8; ++t) {
            const float bias = b2[ch * 128 + t * 16 + row];
            float p = 0.f;
#pragma unroll
            for (int r = 0; r < 8; ++r)
                p += fmaxf(acc[t][r] + bias, 0.f);
            colAcc[t] += p;
        }
    }

    // Reduce lane halves and publish per-wave column sums (128 cols each).
#pragma unroll
    for (int t = 0; t < 8; ++t) {
        float p = colAcc[t];
        p += __shfl_xor(p, 16);          // fold the two lane-halves
        if (hs == 0)
            wsum[wave * 128 + t * 16 + row] = p;   // unique writer
    }
    __syncthreads();

    // Block partial: sum the 4 row strips for this column, write out.
    const int colHalf = tid >> 7;        // 0..1
    const int colLoc  = tid & 127;
    float bp = 0.f;
#pragma unroll
    for (int s = 0; s < 4; ++s)
        bp += wsum[(s * 2 + colHalf) * 128 + colLoc];
    const int blockLinear = (b * 128 + cb) * ITILES + it;
    partials[(size_t)blockLinear * Hh + tid] = bp;
}

// ---------------------------------------------------------------------------
// Tail: acc = Sigma partials; mid = acc@w3 + N^2*b3; o = relu(mid@w4+b4);
// out = o@w5 + b5.  One block per batch, all f32 (cost is trivial).
// ---------------------------------------------------------------------------
__global__ __launch_bounds__(Hh)
void finalize(const float* __restrict__ partials,
              const float* __restrict__ w3, const float* __restrict__ b3,
              const float* __restrict__ w4, const float* __restrict__ b4,
              const float* __restrict__ w5, const float* __restrict__ b5,
              float* __restrict__ out, int blocksPerBatch) {
    __shared__ float accS[Hh], midS[Hh], oS[Hh];
    const int b = blockIdx.x;
    const int h = threadIdx.x;

    float a = 0.f;
    const float* p = partials + (size_t)b * blocksPerBatch * Hh + h;
    for (int j = 0; j < blocksPerBatch; ++j) a += p[(size_t)j * Hh];
    accS[h] = a;
    __syncthreads();

    float m = (float)(Nn) * (float)(Nn) * b3[h];      // N^2 * b3
    for (int k = 0; k < Hh; ++k) m = fmaf(accS[k], w3[(size_t)k * Hh + h], m);
    midS[h] = m;                                      // mid nonlinearity = Identity
    __syncthreads();

    float o4 = b4[h];
    for (int k = 0; k < Hh; ++k) o4 = fmaf(midS[k], w4[(size_t)k * Hh + h], o4);
    oS[h] = fmaxf(o4, 0.f);
    __syncthreads();

    if (h < OUTD) {
        float r = b5[h];
        for (int k = 0; k < Hh; ++k) r = fmaf(oS[k], w5[(size_t)k * OUTD + h], r);
        out[b * OUTD + h] = r;
    }
}

// ---------------------------------------------------------------------------
extern "C" void kernel_launch(void* const* d_in, const int* in_sizes, int n_in,
                              void* d_out, int out_size, void* d_ws, size_t ws_size,
                              hipStream_t stream) {
    (void)in_sizes; (void)n_in; (void)out_size; (void)ws_size;
    const float* input = (const float*)d_in[0];
    const float* w1 = (const float*)d_in[1];
    const float* b1 = (const float*)d_in[2];
    const float* w2 = (const float*)d_in[3];
    const float* b2 = (const float*)d_in[4];
    const float* w3 = (const float*)d_in[5];
    const float* b3 = (const float*)d_in[6];
    const float* w4 = (const float*)d_in[7];
    const float* b4 = (const float*)d_in[8];
    const float* w5 = (const float*)d_in[9];
    const float* b5 = (const float*)d_in[10];
    float* out = (float*)d_out;

    char* ws = (char*)d_ws;
    _Float16* hi   = (_Float16*)(ws);                       // 2*512*256*2 = 512 KB
    _Float16* hc   = (_Float16*)(ws + 524288);              // 512 KB
    _Float16* w2t  = (_Float16*)(ws + 1048576);             // 128 KB
    float*    part = (float*)   (ws + 1179648);             // 2048*256*4 = 2 MB

    prep_hihc<<<dim3(Bn * Nn), dim3(Hh), 0, stream>>>(input, w1, b1, hi, hc);
    prep_w2t <<<dim3(Hh),      dim3(Hh), 0, stream>>>(w2, w2t);

    const size_t ldsBytes = (size_t)Hh * LDS_PITCH * sizeof(_Float16)  // 132 KB w2^T
                          + (size_t)8 * 128 * sizeof(float);           // 4 KB wave sums
    pair_gemm<<<dim3(ITILES, Nn / CPB, Bn), dim3(256), ldsBytes, stream>>>(
        hi, hc, w2t, b2, part);

    const int blocksPerBatch = ITILES * (Nn / CPB);   // 1024
    finalize<<<dim3(Bn), dim3(Hh), 0, stream>>>(part, w3, b3, w4, b4, w5, b5, out,
                                                blocksPerBatch);
}
